// FeatureDenoiser_40913858461930
// MI455X (gfx1250) — matrix operand order repaired and checked by reading, not compiled, not run on hardware
//
#include <hip/hip_runtime.h>
#include <hip/hip_bf16.h>

// ---------------- model constants ----------------
static constexpr int BB   = 8;       // batch
static constexpr int LL   = 2048;    // seq len
static constexpr int DEMB = 768;
static constexpr int DD   = 1024;    // hidden
static constexpr int NS   = 16;      // d_state
static constexpr int RR   = 64;      // dt_rank
static constexpr int NLAY = 6;
static constexpr int MROW = BB * LL; // 16384 rows

// ---------------- WMMA vector types ----------------
typedef __attribute__((ext_vector_type(16))) _Float16 v16h;
typedef __attribute__((ext_vector_type(8)))  _Float16 v8h;
typedef __attribute__((ext_vector_type(8)))  float    v8f;

// ================= GEMM: C = act(A@W + bias) + res1 + res2 =================
// A: (MROW x K) f32, row stride lda. W: (K x N) f32 row-major.
// Block tile 128x128, BK=32, 256 threads = 8 wave32s, wave tile 32x64 (2x4 wmma).
// Double-buffered LDS pipeline: next tile's global loads are issued before the
// current tile's WMMAs so VMEM latency hides under the matrix pipe.
#define BM 128
#define BN 128
#define BKK 32
#define LDSK (BKK + 8)   // 40 halves = 80B row stride (16B aligned)

__global__ __launch_bounds__(256)
void k_gemm_wmma(const float* __restrict__ A, int lda,
                 const float* __restrict__ W, int N, int K,
                 const float* __restrict__ bias,
                 const float* __restrict__ res1,
                 const float* __restrict__ res2,
                 float* __restrict__ C, int ldc, int act)
{
    __shared__ _Float16 As[2][BM][LDSK];
    __shared__ _Float16 Bs[2][BN][LDSK];   // transposed: [n][k]

    const int tid   = threadIdx.x;
    const int lane  = tid & 31;
    const int wv    = tid >> 5;
    const int wy    = wv & 3;           // 4 wave rows  -> M offset wy*32
    const int wx    = wv >> 2;          // 2 wave cols  -> N offset wx*64
    const int tileM = blockIdx.y * BM;
    const int tileN = blockIdx.x * BN;

    const v8f vzero = {0.f,0.f,0.f,0.f,0.f,0.f,0.f,0.f};
    v8f acc[2][4];
#pragma unroll
    for (int mt = 0; mt < 2; ++mt)
#pragma unroll
        for (int nt = 0; nt < 4; ++nt) acc[mt][nt] = vzero;

    // A-load mapping: thread -> (row 0..31 (+rr*32), 4 consecutive k)
    const int arow = tid >> 3;          // 0..31
    const int akq  = (tid & 7) * 4;     // 0..28 step 4
    // B-load mapping: thread -> (n 0..127, k half 0..15 / 16..31)
    const int bn   = tid & 127;
    const int bk0  = (tid >> 7) * 16;   // 0 or 16
    const int ngB  = tileN + bn;
    const bool nok = ngB < N;
    const int ncl  = nok ? ngB : (N - 1);   // clamped in-bounds column

    float4 aReg[4];
    float  bReg[16];

    // issue global loads for the tile starting at k0 (branch-free)
    auto gload = [&](int k0) {
#pragma unroll
        for (int rr = 0; rr < 4; ++rr)
            aReg[rr] = *reinterpret_cast<const float4*>(
                A + (size_t)(tileM + arow + rr * 32) * lda + k0 + akq);
#pragma unroll
        for (int ki = 0; ki < 16; ++ki) {
            const float v = W[(size_t)(k0 + bk0 + ki) * N + ncl];
            bReg[ki] = nok ? v : 0.f;     // v_cndmask, no exec divergence
        }
    };
    // convert f32->f16 and stage registers into LDS buffer `buf`
    auto sstore = [&](int buf) {
#pragma unroll
        for (int rr = 0; rr < 4; ++rr) {
            union { _Float16 hx[4]; uint2 u; } pk;
            pk.hx[0] = (_Float16)aReg[rr].x; pk.hx[1] = (_Float16)aReg[rr].y;
            pk.hx[2] = (_Float16)aReg[rr].z; pk.hx[3] = (_Float16)aReg[rr].w;
            *reinterpret_cast<uint2*>(&As[buf][arow + rr * 32][akq]) = pk.u;
        }
#pragma unroll
        for (int ki = 0; ki < 16; ++ki)
            Bs[buf][bn][bk0 + ki] = (_Float16)bReg[ki];
    };

    // ---- pipeline prologue: stage tile 0 ----
    gload(0);
    sstore(0);
    __syncthreads();

    int cur = 0;
    for (int k0 = 0; k0 < K; k0 += BKK) {
        const bool more = (k0 + BKK) < K;
        if (more) {
            gload(k0 + BKK);              // overlap with WMMAs below
            if (k0 + 2 * BKK < K) {       // warm L2/WGP$ for tile after next
                __builtin_prefetch(A + (size_t)(tileM + arow) * lda + k0 + 2 * BKK + akq, 0, 1);
                __builtin_prefetch(W + (size_t)(k0 + 2 * BKK + bk0) * N + ncl, 0, 1);
            }
        }

        // ---- fragments from LDS[cur] ----
        // A frag: lane holds row m=lane%16; halves 0..7 = K kb..kb+7,
        //         halves 8..15 = K kb+16..kb+23, kb = 8*(lane/16)
        const int akb = (lane >> 4) * 8;
        v16h af[2];
#pragma unroll
        for (int mt = 0; mt < 2; ++mt) {
            const int m = wy * 32 + mt * 16 + (lane & 15);
            const v8h lo = *reinterpret_cast<const v8h*>(&As[cur][m][akb]);
            const v8h hi = *reinterpret_cast<const v8h*>(&As[cur][m][akb + 16]);
            af[mt] = __builtin_shufflevector(lo, hi,
                0,1,2,3,4,5,6,7,8,9,10,11,12,13,14,15);
        }
        // B frag: lane holds col n=lane%16; halves 0..7 = K kb..kb+7,
        //         halves 8..15 = K kb+16..kb+23, kb = 8*(lane/16)
        const int bkb = (lane >> 4) * 8;
#pragma unroll
        for (int nt = 0; nt < 4; ++nt) {
            const int nn = wx * 64 + nt * 16 + (lane & 15);
            const v8h lo = *reinterpret_cast<const v8h*>(&Bs[cur][nn][bkb]);
            const v8h hi = *reinterpret_cast<const v8h*>(&Bs[cur][nn][bkb + 16]);
            const v16h bf = __builtin_shufflevector(lo, hi,
                0,1,2,3,4,5,6,7,8,9,10,11,12,13,14,15);
#pragma unroll
            for (int mt = 0; mt < 2; ++mt) {
                acc[mt][nt] = __builtin_amdgcn_wmma_f32_16x16x32_f16(
                    false, af[mt], false, bf, (short)0, acc[mt][nt], false, false);
            }
        }

        if (more) {
            sstore(cur ^ 1);              // write the other buffer
            __syncthreads();              // one barrier per K-step
            cur ^= 1;
        }
    }

    // ---- epilogue: C/D layout row = r + 8*(lane/16), col = lane%16 ----
    const int rbase = tileM + wy * 32 + 8 * (lane >> 4);
    const int cbase = tileN + wx * 64 + (lane & 15);
#pragma unroll
    for (int mt = 0; mt < 2; ++mt) {
#pragma unroll
        for (int nt = 0; nt < 4; ++nt) {
            const int col = cbase + nt * 16;
            if (col < N) {
                const float bv = bias ? bias[col] : 0.f;
#pragma unroll
                for (int r = 0; r < 8; ++r) {
                    const int row = rbase + mt * 16 + r;
                    float v = acc[mt][nt][r] + bv;
                    if (act == 1)      v = fmaxf(v, 0.f);
                    else if (act == 2) v = (v > 20.f) ? v : log1pf(__expf(v));
                    if (res1) v += res1[(size_t)row * ldc + col];
                    if (res2) v += res2[(size_t)row * ldc + col];
                    C[(size_t)row * ldc + col] = v;
                }
            }
        }
    }
}

// ================= LayerNorm (one block per row of 1024) =================
__global__ __launch_bounds__(256)
void k_layernorm(const float* __restrict__ x, const float* __restrict__ w,
                 const float* __restrict__ b, float* __restrict__ y)
{
    const int row = blockIdx.x;
    const int tid = threadIdx.x;
    const float4 v = *reinterpret_cast<const float4*>(x + (size_t)row * DD + tid * 4);
    float s1 = v.x + v.y + v.z + v.w;
    float s2 = v.x*v.x + v.y*v.y + v.z*v.z + v.w*v.w;
#pragma unroll
    for (int m = 1; m <= 16; m <<= 1) {
        s1 += __shfl_xor(s1, m, 32);
        s2 += __shfl_xor(s2, m, 32);
    }
    __shared__ float r1[8], r2[8];
    if ((tid & 31) == 0) { r1[tid >> 5] = s1; r2[tid >> 5] = s2; }
    __syncthreads();
    s1 = 0.f; s2 = 0.f;
#pragma unroll
    for (int i = 0; i < 8; ++i) { s1 += r1[i]; s2 += r2[i]; }
    const float mean = s1 * (1.f / DD);
    const float var  = s2 * (1.f / DD) - mean * mean;
    const float rs   = rsqrtf(var + 1e-5f);
    const float4 wv = *reinterpret_cast<const float4*>(w + tid * 4);
    const float4 bv = *reinterpret_cast<const float4*>(b + tid * 4);
    float4 o;
    o.x = (v.x - mean) * rs * wv.x + bv.x;
    o.y = (v.y - mean) * rs * wv.y + bv.y;
    o.z = (v.z - mean) * rs * wv.z + bv.z;
    o.w = (v.w - mean) * rs * wv.w + bv.w;
    *reinterpret_cast<float4*>(y + (size_t)row * DD + tid * 4) = o;
}

// ================= RMSNorm =================
__global__ __launch_bounds__(256)
void k_rmsnorm(const float* __restrict__ x, const float* __restrict__ w,
               float* __restrict__ y)
{
    const int row = blockIdx.x;
    const int tid = threadIdx.x;
    const float4 v = *reinterpret_cast<const float4*>(x + (size_t)row * DD + tid * 4);
    float s2 = v.x*v.x + v.y*v.y + v.z*v.z + v.w*v.w;
#pragma unroll
    for (int m = 1; m <= 16; m <<= 1) s2 += __shfl_xor(s2, m, 32);
    __shared__ float r2[8];
    if ((tid & 31) == 0) r2[tid >> 5] = s2;
    __syncthreads();
    s2 = 0.f;
#pragma unroll
    for (int i = 0; i < 8; ++i) s2 += r2[i];
    const float rs = rsqrtf(s2 * (1.f / DD) + 1e-5f);
    const float4 wv = *reinterpret_cast<const float4*>(w + tid * 4);
    float4 o;
    o.x = v.x * rs * wv.x; o.y = v.y * rs * wv.y;
    o.z = v.z * rs * wv.z; o.w = v.w * rs * wv.w;
    *reinterpret_cast<float4*>(y + (size_t)row * DD + tid * 4) = o;
}

// ======= causal depthwise conv (K=4) + SiLU; input is xm half of xz (ld 2D) =======
__global__ __launch_bounds__(256)
void k_conv_silu(const float* __restrict__ xz, const float* __restrict__ cw,
                 const float* __restrict__ cb, float* __restrict__ out)
{
    const int id = blockIdx.x * 256 + threadIdx.x;   // over MROW*DD
    const int d   = id & (DD - 1);
    const int row = id >> 10;
    const int t   = row & (LL - 1);
    const int rb  = row - t;                          // b*L
    float acc = cb[d];
#pragma unroll
    for (int j = 0; j < 4; ++j) {
        const int tt = t - 3 + j;
        if (tt >= 0)
            acc += xz[((size_t)(rb + tt)) * (2 * DD) + d] * cw[d * 4 + j];
    }
    out[(size_t)id] = acc / (1.f + __expf(-acc));     // SiLU
}

// ======= selective scan: one lane per (b,d,n); n in consecutive lanes =======
__global__ __launch_bounds__(256)
void k_scan(const float* __restrict__ dlt, const float* __restrict__ xm,
            const float* __restrict__ dbc, const float* __restrict__ A_log,
            float* __restrict__ ys)
{
    const int g = blockIdx.x * 256 + threadIdx.x;     // B*D*N = 131072 lanes
    const int n = g & 15;
    const int d = (g >> 4) & (DD - 1);
    const int b = g >> 14;
    const float Adn = -__expf(A_log[d * NS + n]);     // A = -exp(A_log)

    const float* dl = dlt + (size_t)b * LL * DD + d;
    const float* xp = xm  + (size_t)b * LL * DD + d;
    const float* Bp = dbc + (size_t)b * LL * 96 + RR + n;        // Bm
    const float* Cp = dbc + (size_t)b * LL * 96 + RR + NS + n;   // Cm
    float* yp = ys + (size_t)b * LL * DD + d;

    float hst = 0.f;
    for (int t = 0; t < LL; ++t) {
        const float dt = dl[(size_t)t * DD];
        const float xt = xp[(size_t)t * DD];
        const float Bt = Bp[(size_t)t * 96];
        const float Ct = Cp[(size_t)t * 96];
        hst = __expf(dt * Adn) * hst + (dt * xt) * Bt;
        float s = hst * Ct;
        s += __shfl_xor(s, 1, 32);
        s += __shfl_xor(s, 2, 32);
        s += __shfl_xor(s, 4, 32);
        s += __shfl_xor(s, 8, 32);
        if (n == 0) yp[(size_t)t * DD] = s;
    }
}

// ======= gate: t0 = (ys + xm*Dp) * silu(z)  (z = second half of xz) =======
__global__ __launch_bounds__(256)
void k_gate(const float* __restrict__ ys, const float* __restrict__ xm,
            const float* __restrict__ Dp, const float* __restrict__ xz,
            float* __restrict__ out)
{
    const int id = blockIdx.x * 256 + threadIdx.x;   // over MROW*DD
    const int d   = id & (DD - 1);
    const int row = id >> 10;
    const float z = xz[(size_t)row * (2 * DD) + DD + d];
    const float v = (ys[id] + xm[id] * Dp[d]) * (z / (1.f + __expf(-z)));
    out[id] = v;
}

// =================================================================
static inline int cdiv_i(int a, int b) { return (a + b - 1) / b; }

extern "C" void kernel_launch(void* const* d_in, const int* in_sizes, int n_in,
                              void* d_out, int out_size, void* d_ws, size_t ws_size,
                              hipStream_t stream)
{
    (void)in_sizes; (void)n_in; (void)out_size; (void)ws_size;

    const float* x     = (const float*)d_in[0];
    const float* w_in  = (const float*)d_in[1];
    const float* b_in  = (const float*)d_in[2];
    const float* w_out = (const float*)d_in[3];
    const float* b_out = (const float*)d_in[4];
    const float* ln1w  = (const float*)d_in[5];
    const float* ln1b  = (const float*)d_in[6];
    const float* ln2w  = (const float*)d_in[7];
    const float* ln2b  = (const float*)d_in[8];
    const float* rmsw  = (const float*)d_in[9];
    const float* iw    = (const float*)d_in[10];
    const float* cw    = (const float*)d_in[11];
    const float* cb    = (const float*)d_in[12];
    const float* xw    = (const float*)d_in[13];
    const float* dw    = (const float*)d_in[14];
    const float* db    = (const float*)d_in[15];
    const float* Alog  = (const float*)d_in[16];
    const float* Dp    = (const float*)d_in[17];
    const float* ow    = (const float*)d_in[18];
    const float* w1    = (const float*)d_in[19];
    const float* b1    = (const float*)d_in[20];
    const float* w2    = (const float*)d_in[21];
    const float* b2    = (const float*)d_in[22];

    // ---- workspace layout (all sizes multiples of 256B) ----
    char* ws = (char*)d_ws;
    size_t off = 0;
    auto alloc = [&](size_t bytes) -> float* {
        float* p = (float*)(ws + off);
        off += (bytes + 255) & ~(size_t)255;
        return p;
    };
    const size_t SZ_MD = (size_t)MROW * DD * sizeof(float);       // 64 MiB
    float* h    = alloc(SZ_MD);
    float* hn   = alloc(SZ_MD);
    float* t0   = alloc(SZ_MD);
    float* xz   = alloc((size_t)MROW * 2 * DD * sizeof(float));   // 128 MiB
    float* xm   = alloc(SZ_MD);
    float* dbc  = alloc((size_t)MROW * 96 * sizeof(float));
    float* dlt  = alloc(SZ_MD);
    float* ysum = alloc(SZ_MD);
    float* mid  = xz;  // FFN mid (MROW x 4D = 256 MiB) aliases dead xz/xm/dbc/dlt region

    auto gemm = [&](const float* A, int lda, const float* W, int N, int K,
                    const float* bias, const float* r1, const float* r2,
                    float* C, int ldc, int act) {
        dim3 grid(cdiv_i(N, BN), MROW / BM);
        k_gemm_wmma<<<grid, 256, 0, stream>>>(A, lda, W, N, K, bias, r1, r2, C, ldc, act);
    };

    const int NELEM_BLK = (MROW * DD) / 256;   // 65536 blocks for elementwise

    // ---- embed: h = x @ w_in + b_in ----
    gemm(x, DEMB, w_in, DD, DEMB, b_in, nullptr, nullptr, h, DD, 0);

    for (int i = 0; i < NLAY; ++i) {
        // Mamba residual block
        k_layernorm<<<MROW, 256, 0, stream>>>(h, ln1w + i * DD, ln1b + i * DD, hn);
        k_rmsnorm  <<<MROW, 256, 0, stream>>>(hn, rmsw + i * DD, t0);
        gemm(t0, DD, iw + (size_t)i * DD * 2 * DD, 2 * DD, DD,
             nullptr, nullptr, nullptr, xz, 2 * DD, 0);
        k_conv_silu<<<NELEM_BLK, 256, 0, stream>>>(xz, cw + (size_t)i * DD * 4,
                                                   cb + i * DD, xm);
        gemm(xm, DD, xw + (size_t)i * DD * 96, 96, DD,
             nullptr, nullptr, nullptr, dbc, 96, 0);
        gemm(dbc, 96, dw + (size_t)i * RR * DD, DD, RR,
             db + i * DD, nullptr, nullptr, dlt, DD, 2 /*softplus*/);
        k_scan<<<(BB * DD * NS) / 256, 256, 0, stream>>>(dlt, xm, dbc,
                                                         Alog + (size_t)i * DD * NS, ysum);
        k_gate<<<NELEM_BLK, 256, 0, stream>>>(ysum, xm, Dp + i * DD, xz, t0);
        // h = h + hn + t0 @ ow   (mamba out + inner residual + outer residual)
        gemm(t0, DD, ow + (size_t)i * DD * DD, DD, DD,
             nullptr, h, hn, h, DD, 0);

        // FFN block
        k_layernorm<<<MROW, 256, 0, stream>>>(h, ln2w + i * DD, ln2b + i * DD, hn);
        gemm(hn, DD, w1 + (size_t)i * DD * 4 * DD, 4 * DD, DD,
             b1 + (size_t)i * 4 * DD, nullptr, nullptr, mid, 4 * DD, 1 /*relu*/);
        gemm(mid, 4 * DD, w2 + (size_t)i * 4 * DD * DD, DD, 4 * DD,
             b2 + i * DD, h, nullptr, h, DD, 0);
    }

    // ---- head: out = h @ w_out + b_out ----
    gemm(h, DD, w_out, DEMB, DD, b_out, nullptr, nullptr, (float*)d_out, DEMB, 0);
}
